// MPS_20564303413829
// MI455X (gfx1250) — matrix operand order repaired and checked by reading, not compiled, not run on hardware
//
#include <hip/hip_runtime.h>
#include <hip/hip_bf16.h>

typedef _Float16 v16h __attribute__((ext_vector_type(16)));
typedef float    v8f  __attribute__((ext_vector_type(8)));

#define N_SITES 128
#define N_BULK  126
#define BOND    64
#define WAVES_PER_BLOCK 4
#define EPSF 1e-30f

// ---------------------------------------------------------------------------
// Prep: swizzle bulk_tensors (f32, [126][64][2][64]) into WMMA B-fragment
// layout, f16, stacked over phys: B'[k,n], k = phys*64 + bond_in (K=128).
// Fragment (site, jt, kc): lane ln holds col n = jt*16 + (ln&15),
// K = kc*32 + ((ln&16)?8:0) + e + (e>=8?8:0), e = 0..15, stored lane-contiguous (32B/lane).
// ---------------------------------------------------------------------------
__global__ __launch_bounds__(256) void mps_swizzleB(const float* __restrict__ bulk,
                                                    _Float16* __restrict__ bsw) {
  int t = blockIdx.x * blockDim.x + threadIdx.x;     // one thread per (site,jt,kc,lane)
  const int total = N_BULK * 4 * 4 * 32;
  if (t >= total) return;
  int lane = t & 31;
  int kc   = (t >> 5) & 3;
  int jt   = (t >> 7) & 3;
  int site = t >> 9;
  int n     = jt * 16 + (lane & 15);
  int kbase = kc * 32 + ((lane & 16) ? 8 : 0);
  v16h vals;
#pragma unroll
  for (int e = 0; e < 16; ++e) {
    int k = kbase + e + ((e >= 8) ? 8 : 0);  // 0..127 (stacked K)
    int p = k >> 6;            // phys select
    int i = k & 63;            // bond_in
    vals[e] = (_Float16)bulk[((size_t)(site * 64 + i) * 2 + p) * 64 + n];
  }
  *(v16h*)(bsw + (size_t)t * 16) = vals;
}

// ---------------------------------------------------------------------------
// Main: 1 wave handles 16 samples through the whole chain.
// env tile (16x64) lives in LDS in A-fragment f16 layout (2 chunks of K=0..63).
// Per site: D(16x64) = A'(16x128, per-lane zero-selected) x B'(128x64).
// ---------------------------------------------------------------------------
__global__ __launch_bounds__(32 * WAVES_PER_BLOCK) void mps_chain_kernel(
    const int* __restrict__ cfg, const float* __restrict__ left,
    const _Float16* __restrict__ bsw, const float* __restrict__ right,
    float* __restrict__ out) {
  // per-wave env fragments: [kc(2)][lane'(32)][e(16)] f16
  __shared__ alignas(32) _Float16 envFrag[WAVES_PER_BLOCK][2][32][16];
  __shared__ float lsInit[WAVES_PER_BLOCK][16];
  __shared__ float lsFinal[WAVES_PER_BLOCK][16];

  const int tid   = threadIdx.x;
  const int wave  = tid >> 5;
  const int lane  = tid & 31;
  const int h     = lane >> 4;        // wave half: rows 8h..8h+7 in C/D layout
  const int ln16  = lane & 15;
  const int tile  = blockIdx.x * WAVES_PER_BLOCK + wave;
  const int mbase = tile * 16;

  // ---- init env from left tensor (lanes 0..15, one sample row each) ----
  if (lane < 16) {
    const int m = mbase + lane;
    const float* lrow = left + cfg[(size_t)m * N_SITES] * BOND;
    float mx = EPSF;
#pragma unroll 8
    for (int k = 0; k < BOND; ++k) mx = fmaxf(mx, fabsf(lrow[k]));
    const float inv = 1.0f / mx;
    lsInit[wave][lane] = logf(mx);
#pragma unroll 8
    for (int k = 0; k < BOND; ++k) {
      int kc = k >> 5, kk = k & 31;
      int hh = (kk >> 3) & 1;
      int e  = ((kk >> 4) << 3) | (kk & 7);
      envFrag[wave][kc][lane + 16 * hh][e] = (_Float16)(lrow[k] * inv);
    }
  }
  __syncthreads();

  float ls8[8];
#pragma unroll
  for (int r = 0; r < 8; ++r) ls8[r] = lsInit[wave][r + 8 * h];

  const v16h zero = {};

  for (int s = 0; s < N_BULK; ++s) {
    // --- load env A-chunks (K 0..63) from LDS, lane-contiguous ---
    v16h a0 = *(const v16h*)&envFrag[wave][0][lane][0];
    v16h a1 = *(const v16h*)&envFrag[wave][1][lane][0];
    // per-sample phys selection for this site (row M = lane&15)
    const int v = cfg[(size_t)(mbase + ln16) * N_SITES + (s + 1)];
    v16h ae0 = v ? zero : a0;   // K   0..31 : phys-0 slice
    v16h ae1 = v ? zero : a1;   // K  32..63
    v16h ae2 = v ? a0 : zero;   // K  64..95 : phys-1 slice
    v16h ae3 = v ? a1 : zero;   // K 96..127

    const _Float16* sb = bsw + (size_t)s * (4 * 4 * 32 * 16);
    if (s + 1 < N_BULK) {
      __builtin_prefetch(sb + 4 * 4 * 32 * 16 + lane * 256, 0, 0);
      __builtin_prefetch(sb + 4 * 4 * 32 * 16 + 8192 + lane * 256, 0, 0);
    }

    v8f acc[4];
#pragma unroll
    for (int jt = 0; jt < 4; ++jt) {
      const _Float16* fb = sb + (size_t)(jt * 4) * (32 * 16) + lane * 16;
      v16h b0 = *(const v16h*)(fb);
      v16h b1 = *(const v16h*)(fb + 512);
      v16h b2 = *(const v16h*)(fb + 1024);
      v16h b3 = *(const v16h*)(fb + 1536);
      v8f c = {};
      c = __builtin_amdgcn_wmma_f32_16x16x32_f16(false, ae0, false, b0, (short)0, c, false, false);
      c = __builtin_amdgcn_wmma_f32_16x16x32_f16(false, ae1, false, b1, (short)0, c, false, false);
      c = __builtin_amdgcn_wmma_f32_16x16x32_f16(false, ae2, false, b2, (short)0, c, false, false);
      c = __builtin_amdgcn_wmma_f32_16x16x32_f16(false, ae3, false, b3, (short)0, c, false, false);
      acc[jt] = c;
    }

    // --- per-row max-abs: reduce over the 4 col-tiles, then butterfly over
    //     the 16 lanes of each half (rows r+8h live in VGPR r of half h) ---
    float inv8[8];
#pragma unroll
    for (int r = 0; r < 8; ++r) {
      float t = fmaxf(fmaxf(fabsf(acc[0][r]), fabsf(acc[1][r])),
                      fmaxf(fabsf(acc[2][r]), fabsf(acc[3][r])));
      t = fmaxf(t, __shfl_xor(t, 1, 32));
      t = fmaxf(t, __shfl_xor(t, 2, 32));
      t = fmaxf(t, __shfl_xor(t, 4, 32));
      t = fmaxf(t, __shfl_xor(t, 8, 32));
      t = fmaxf(t, EPSF);
      ls8[r] += logf(t);
      inv8[r] = 1.0f / t;
    }

    // --- scale, convert, scatter new env into A-fragment layout ---
    // value at (VGPR r, lane) is (M = r + 8h, K = jt*16 + ln16)
#pragma unroll
    for (int jt = 0; jt < 4; ++jt) {
      const int K  = jt * 16 + ln16;
      const int kc = K >> 5, kk = K & 31;
      const int hh = (kk >> 3) & 1;
      const int e  = ((kk >> 4) << 3) | (kk & 7);
      _Float16* base = &envFrag[wave][kc][8 * h + 16 * hh][e];
#pragma unroll
      for (int r = 0; r < 8; ++r)
        base[(size_t)r * 16] = (_Float16)(acc[jt][r] * inv8[r]);
    }
  }

  // ---- publish log-scales (each half's lane 0 holds its 8 rows) ----
  if (lane == 0) {
#pragma unroll
    for (int r = 0; r < 8; ++r) lsFinal[wave][r] = ls8[r];
  }
  if (lane == 16) {
#pragma unroll
    for (int r = 0; r < 8; ++r) lsFinal[wave][8 + r] = ls8[r];
  }
  __syncthreads();

  // ---- right boundary contraction + output (lanes 0..15) ----
  if (lane < 16) {
    const int m = mbase + lane;
    const int v = cfg[(size_t)m * N_SITES + (N_SITES - 1)];
    float psi = 0.0f;
#pragma unroll 8
    for (int k = 0; k < BOND; ++k) {
      int kc = k >> 5, kk = k & 31;
      int hh = (kk >> 3) & 1;
      int e  = ((kk >> 4) << 3) | (kk & 7);
      float ev = (float)envFrag[wave][kc][lane + 16 * hh][e];
      psi += ev * right[k * 2 + v];
    }
    float am = fmaxf(fabsf(psi), EPSF);
    out[m] = 2.0f * (lsFinal[wave][lane] + logf(am));
  }
}

extern "C" void kernel_launch(void* const* d_in, const int* in_sizes, int n_in,
                              void* d_out, int out_size, void* d_ws, size_t ws_size,
                              hipStream_t stream) {
  const int*   cfg   = (const int*)d_in[0];    // (BATCH, 128) int32
  const float* left  = (const float*)d_in[1];  // (1, 2, 64)
  const float* bulk  = (const float*)d_in[2];  // (126, 64, 2, 64)
  const float* right = (const float*)d_in[3];  // (64, 2, 1)
  float* out = (float*)d_out;

  const int batch = in_sizes[0] / N_SITES;     // 8192

  _Float16* bsw = (_Float16*)d_ws;             // ~2 MB swizzled f16 B-fragments

  const int prepThreads = N_BULK * 4 * 4 * 32; // 64512
  mps_swizzleB<<<(prepThreads + 255) / 256, 256, 0, stream>>>(bulk, bsw);

  const int blocks = batch / (16 * WAVES_PER_BLOCK); // 128
  mps_chain_kernel<<<blocks, 32 * WAVES_PER_BLOCK, 0, stream>>>(cfg, left, bsw, right, out);
}
